// GLABlock_11991548690954
// MI455X (gfx1250) — matrix operand order repaired and checked by reading, compiled, not run on hardware
//
#include <hip/hip_runtime.h>

// ---------------------------------------------------------------------------
// GLA block for MI455X (gfx1250, wave32, WMMA bf16 16x16x32 + TDM).
// B=4, T=2048, D=1024, H=4, DK=DV=256, chunk C=64.
// - All GEMMs run through v_wmma_f32_16x16x32_bf16 (fp32 accumulate).
// - GEMM weight block-tiles are DMA'd to LDS with tensor_load_to_lds
//   (Tensor Data Mover, TENSORcnt), double-buffered across K-slabs.
// - The GLA recurrence is chunkwise: per (b,h) workgroup the 256x256 state
//   lives in LDS (~265KB dynamic -> needs CDNA5's 320KB/WGP LDS).
// ---------------------------------------------------------------------------

#define GB 4
#define GT 2048
#define GD 1024
#define GH 4
#define GDK 256
#define GDV 256
#define GC 64
#define GM (GB * GT)          // 8192 rows
#define GNCHUNK (GT / GC)     // 32
#define KS 256                // GEMM K-slab staged per TDM transfer

typedef __attribute__((ext_vector_type(16))) __bf16 v16bf;
typedef __attribute__((ext_vector_type(8)))  __bf16 v8bf;
typedef __attribute__((ext_vector_type(8)))  float  v8f;
typedef __attribute__((ext_vector_type(4)))  int    v4i;
typedef __attribute__((ext_vector_type(8)))  int    v8i;

__device__ __forceinline__ v16bf cat8(v8bf lo, v8bf hi) {
  return __builtin_shufflevector(lo, hi, 0,1,2,3,4,5,6,7,8,9,10,11,12,13,14,15);
}

// A fragment (16x32 bf16): row-major source, row stride ld, tile row m0.
// Lanes 0-15 hold row M=lane, K = {0..7, 16..23}; lanes 16-31: K = {8..15, 24..31}.
__device__ __forceinline__ v16bf load_frag_a(const __bf16* base, int ld, int m0,
                                             int k0, int lane) {
  const __bf16* rowp = base + (size_t)(m0 + (lane & 15)) * ld + k0;
  const int h8 = ((lane >> 4) & 1) * 8;
  v8bf lo = *(const v8bf*)(rowp + h8);
  v8bf hi = *(const v8bf*)(rowp + 16 + h8);
  return cat8(lo, hi);
}

// B fragment (32x16 bf16) read from a transposed (src[n][k], row stride ld)
// buffer: lane n = n0+lane%16 holds K = {0..7, 16..23} (lanes 0-15) or {8..15, 24..31} (lanes 16-31).
__device__ __forceinline__ v16bf load_frag_bt(const __bf16* base, int ld, int n0,
                                              int k0, int lane) {
  const __bf16* colp = base + (size_t)(n0 + (lane & 15)) * ld + k0
                            + ((lane >> 4) & 1) * 8;
  v8bf lo = *(const v8bf*)(colp);
  v8bf hi = *(const v8bf*)(colp + 16);
  return cat8(lo, hi);
}

__device__ __forceinline__ v8f wmma_bf16(v16bf a, v16bf b, v8f c) {
  return __builtin_amdgcn_wmma_f32_16x16x32_bf16(false, a, false, b,
                                                 (short)0, c, false, false);
}

// ---------------------------------------------------------------------------
// Tensor Data Mover: DMA a 2D bf16 tile [rows][ks] (row stride K elements)
// from global memory into LDS at byte offset lds_off. Scalar (wave-level) op,
// tracked with TENSORcnt. D# built per CDNA5 ISA.
// ---------------------------------------------------------------------------
__device__ __forceinline__ void tdm_load_tile(const __bf16* gsrc, unsigned lds_off,
                                              int rows, int ks, int K) {
  const unsigned long long ga = (unsigned long long)(uintptr_t)gsrc;
  v4i g0;
  g0[0] = 1;                                           // count=1 (valid user D#)
  g0[1] = (int)lds_off;                                // lds_addr
  g0[2] = (int)(unsigned)(ga & 0xFFFFFFFFull);         // global_addr[31:0]
  g0[3] = (int)((unsigned)((ga >> 32) & 0x1FFFFFFull)  // global_addr[56:32]
              | (2u << 30));                           // type=2 ("image")
  v8i g1;
  g1[0] = 0x00010000;                                  // data_size=1 (2B), mask=0
  g1[1] = (K & 0xFFFF) << 16;                          // tensor_dim0[15:0]
  g1[2] = ((K >> 16) & 0xFFFF) | ((rows & 0xFFFF) << 16); // dim0 hi | tensor_dim1 lo
  g1[3] = ((ks & 0xFFFF) << 16);                       // tensor_dim1 hi=0 | tile_dim0
  g1[4] = (rows & 0xFFFF);                             // tile_dim1 | tile_dim2=0
  g1[5] = K;                                           // tensor_dim0_stride[31:0]
  g1[6] = 0;                                           // stride hi | dim1_stride lo
  g1[7] = 0;
  asm volatile("tensor_load_to_lds %0, %1" :: "s"(g0), "s"(g1) : "memory");
}

__device__ __forceinline__ float block_reduce_sum256(float v, float* red) {
  const int t = threadIdx.x;
  red[t] = v;
  __syncthreads();
  for (int off = 128; off > 0; off >>= 1) {
    if (t < off) red[t] += red[t + off];
    __syncthreads();
  }
  float r = red[0];
  __syncthreads();
  return r;
}

// ---------------------------------------------------------------------------
// LayerNorm (fp32 in) -> bf16 out. One 256-thread block per row of D=1024.
// ---------------------------------------------------------------------------
__global__ __launch_bounds__(256)
void layernorm_bf16_kernel(const float* __restrict__ x,
                           const float* __restrict__ w,
                           const float* __restrict__ b,
                           __bf16* __restrict__ out) {
  __shared__ float red[256];
  const size_t m = blockIdx.x;
  const float* xr = x + m * GD;
  float s = 0.f;
  for (int i = threadIdx.x; i < GD; i += 256) s += xr[i];
  const float mu = block_reduce_sum256(s, red) * (1.0f / GD);
  float v = 0.f;
  for (int i = threadIdx.x; i < GD; i += 256) {
    float d = xr[i] - mu;
    v += d * d;
  }
  const float var = block_reduce_sum256(v, red) * (1.0f / GD);
  const float inv = rsqrtf(var + 1e-5f);
  for (int i = threadIdx.x; i < GD; i += 256)
    out[m * GD + i] = (__bf16)((xr[i] - mu) * inv * w[i] + b[i]);
}

// ---------------------------------------------------------------------------
// Transpose + cast: W[K][N] fp32 -> WT[N][K] bf16 (so WMMA B-fragments are
// contiguous 16B loads along K and TDM tiles are row-major over K).
// ---------------------------------------------------------------------------
__global__ __launch_bounds__(256)
void transpose_cast_kernel(const float* __restrict__ W, __bf16* __restrict__ WT,
                           int K, int N) {
  size_t idx = (size_t)blockIdx.x * 256 + threadIdx.x;
  if (idx >= (size_t)K * N) return;
  int k = (int)(idx / N);
  int n = (int)(idx % N);
  WT[(size_t)n * K + k] = (__bf16)W[idx];
}

// ---------------------------------------------------------------------------
// Low-rank forget gate: gk = logsigmoid((xn @ Wgk1) @ Wgk2 + b) / 16.
// ---------------------------------------------------------------------------
__global__ __launch_bounds__(256)
void gate_lowrank_kernel(const __bf16* __restrict__ xn,
                         const float* __restrict__ Wgk1,
                         const float* __restrict__ Wgk2,
                         const float* __restrict__ bgk2,
                         float* __restrict__ gkout) {
  __shared__ float red[256];
  __shared__ float t16[16];
  const size_t m = blockIdx.x;
  float xv[4];
#pragma unroll
  for (int i = 0; i < 4; ++i)
    xv[i] = (float)xn[m * GD + threadIdx.x + i * 256];
  for (int l = 0; l < 16; ++l) {
    float p = 0.f;
#pragma unroll
    for (int i = 0; i < 4; ++i) {
      int d = threadIdx.x + i * 256;
      p += xv[i] * Wgk1[(size_t)d * 16 + l];
    }
    float tot = block_reduce_sum256(p, red);
    if (threadIdx.x == 0) t16[l] = tot;
    __syncthreads();
  }
#pragma unroll
  for (int i = 0; i < 4; ++i) {
    int n = threadIdx.x + i * 256;
    float a = bgk2[n];
#pragma unroll
    for (int l = 0; l < 16; ++l) a += t16[l] * Wgk2[(size_t)l * GD + n];
    float ls = fminf(a, 0.f) - log1pf(__expf(-fabsf(a)));   // stable log_sigmoid
    gkout[m * GD + n] = ls * (1.0f / 16.0f);
  }
}

// ---------------------------------------------------------------------------
// WMMA GEMM with TDM weight staging:
//   C[M][N] = act(A_bf[M][K] @ WT_bf[N][K]^T + bias) (+res)
// 256 threads = 8 waves; block tile 128(M) x 64(N). The 64xKS weight slab is
// DMA'd into LDS by wave 0 via tensor_load_to_lds, double-buffered across
// K-slabs; compute reads B fragments from LDS, A fragments stream from global.
// ---------------------------------------------------------------------------
template <int ACT, bool HAS_BIAS, bool HAS_RES, bool STORE_F32, bool STORE_BF>
__global__ __launch_bounds__(256)
void gemm_bf16_kernel(const __bf16* __restrict__ A, const __bf16* __restrict__ WT,
                      const float* __restrict__ bias, const float* __restrict__ res,
                      float* __restrict__ C, __bf16* __restrict__ Cbf,
                      int N, int K) {
  __shared__ __bf16 Bs[2][64 * KS];                    // 64 KB double buffer
  const int wave = threadIdx.x >> 5;
  const int lane = threadIdx.x & 31;
  const int m0 = blockIdx.x * 128 + wave * 16;
  const int n0 = blockIdx.y * 64;
  const __bf16* arow = A + (size_t)(m0 + (lane & 15)) * K;
  const int nslab = K / KS;
  v8f acc[4] = {};

  if (wave == 0)                                       // prologue: slab 0 DMA
    tdm_load_tile(WT + (size_t)n0 * K, (unsigned)(uintptr_t)&Bs[0][0], 64, KS, K);

  for (int s = 0; s < nslab; ++s) {
    if (wave == 0) {
      if (s + 1 < nslab) {                             // overlap next slab DMA
        tdm_load_tile(WT + (size_t)n0 * K + (s + 1) * KS,
                      (unsigned)(uintptr_t)&Bs[(s + 1) & 1][0], 64, KS, K);
        __builtin_amdgcn_s_wait_tensorcnt(1);          // slab s complete
      } else {
        __builtin_amdgcn_s_wait_tensorcnt(0);          // final slab complete
      }
    }
    __syncthreads();                                   // publish slab s
    const __bf16* bbuf = &Bs[s & 1][0];
    const int kbase = s * KS;
#pragma unroll 4
    for (int k0 = 0; k0 < KS; k0 += 32) {
      __builtin_prefetch(arow + kbase + k0 + 256, 0, 1);  // global_prefetch_b8
      const int h8 = ((lane >> 4) & 1) * 8;
      v8bf alo = *(const v8bf*)(arow + kbase + k0 + h8);
      v8bf ahi = *(const v8bf*)(arow + kbase + k0 + 16 + h8);
      v16bf a = cat8(alo, ahi);
#pragma unroll
      for (int j = 0; j < 4; ++j) {
        v16bf bb = load_frag_bt(bbuf, KS, j * 16, k0, lane);
        acc[j] = wmma_bf16(a, bb, acc[j]);
      }
    }
    __syncthreads();                                   // done with slab s buffer
  }

  const int r0 = m0 + ((lane >> 4) << 3);
#pragma unroll
  for (int j = 0; j < 4; ++j) {
    const int col = n0 + j * 16 + (lane & 15);
    const float bv = HAS_BIAS ? bias[col] : 0.0f;
#pragma unroll
    for (int r = 0; r < 8; ++r) {
      size_t idx = (size_t)(r0 + r) * N + col;
      float v = acc[j][r] + bv;
      if (ACT == 1) { float t = v > 0.0f ? v : 0.01f * v; v = t * t; }  // lrelu^2
      if (HAS_RES) v += res[idx];
      if (STORE_F32) C[idx] = v;
      if (STORE_BF) Cbf[idx] = (__bf16)v;
    }
  }
}

// ---------------------------------------------------------------------------
// Chunkwise GLA scan. One workgroup per (b,h). State S^T [dv][dk] bf16 in LDS.
// Per chunk of 64 tokens, with inclusive log-gate cumsum G and Gend=G[63]:
//   qh = q * exp(G) * DK^-0.5
//   kh = k * exp(-G); kc^T = kh^T * exp(Gend)
//   P  = causal(qh @ kh^T)             (WMMA, K=256)
//   o  = qh @ S + P @ v                (WMMA, K=256 and K=64)
//   S  = S .* exp(Gend)[dk] + kc^T @ v (WMMA, K=64)
// ---------------------------------------------------------------------------
__global__ __launch_bounds__(256)
void gla_scan_kernel(const float* __restrict__ qkvg,
                     const float* __restrict__ gk,
                     float* __restrict__ obuf) {
  extern __shared__ char smem[];
  __bf16* ST  = (__bf16*)smem;            // [DV][DK]  128 KB (state, transposed)
  __bf16* QH  = ST + GDV * GDK;           // [C][DK]   32 KB
  __bf16* KH  = QH + GC * GDK;            // [C][DK]   32 KB
  __bf16* KCT = KH + GC * GDK;            // [DK][C]   32 KB (kc^T)
  __bf16* VT  = KCT + GDK * GC;           // [DV][C]   32 KB (v^T)
  __bf16* Pm  = VT + GDV * GC;            // [C][C]    8 KB
  float*  GE  = (float*)(Pm + GC * GC);   // [DK]      1 KB

  const int bh = blockIdx.x;
  const int b = bh / GH, h = bh % GH;
  const int tid = threadIdx.x;
  const int wave = tid >> 5, lane = tid & 31;

  for (int i = tid; i < GDV * GDK; i += 256) ST[i] = (__bf16)0.0f;
  __syncthreads();

  for (int ck = 0; ck < GNCHUNK; ++ck) {
    // ---- phase 1: load chunk, build gated operands (thread = one d column)
    {
      const int d = tid;
      float g = 0.f;
      for (int t = 0; t < GC; ++t) {
        const size_t m = (size_t)(b * GT + ck * GC + t);
        const size_t base = m * 4096;
        const float qv = qkvg[base + h * 256 + d];
        const float kv = qkvg[base + 1024 + h * 256 + d];
        const float vv = qkvg[base + 2048 + h * 256 + d];
        g += gk[m * GD + h * 256 + d];
        QH[t * GDK + d] = (__bf16)(qv * __expf(g) * 0.0625f);   // DK^-0.5
        const float khv = kv * __expf(-g);
        KH[t * GDK + d] = (__bf16)khv;
        KCT[d * GC + t] = (__bf16)khv;   // rescaled by exp(Gend) below
        VT[d * GC + t]  = (__bf16)vv;
      }
      GE[d] = g;
      const float eg = __expf(g);
      for (int t = 0; t < GC; ++t)
        KCT[d * GC + t] = (__bf16)((float)KCT[d * GC + t] * eg);
    }
    __syncthreads();

    // ---- phase 2: P = causal(QH @ KH^T), 16 tiles / 8 waves
#pragma unroll
    for (int i = 0; i < 2; ++i) {
      const int tt = wave + i * 8;
      const int tm = tt >> 2, tn = tt & 3;
      v8f acc = {};
#pragma unroll
      for (int k0 = 0; k0 < GDK; k0 += 32) {
        v16bf a  = load_frag_a(QH, GDK, tm * 16, k0, lane);
        v16bf bb = load_frag_bt(KH, GDK, tn * 16, k0, lane);
        acc = wmma_bf16(a, bb, acc);
      }
      const int trow0 = tm * 16 + ((lane >> 4) << 3);
      const int scol  = tn * 16 + (lane & 15);
#pragma unroll
      for (int r = 0; r < 8; ++r) {
        const int trow = trow0 + r;
        Pm[trow * GC + scol] = (scol <= trow) ? (__bf16)acc[r] : (__bf16)0.0f;
      }
    }
    __syncthreads();

    // ---- phase 3: o = QH @ S + P @ v, 64 tiles / 8 waves
    for (int i = 0; i < 8; ++i) {
      const int tt = wave * 8 + i;
      const int tm = tt >> 4, tn = tt & 15;
      v8f acc = {};
#pragma unroll 4
      for (int k0 = 0; k0 < GDK; k0 += 32) {
        v16bf a  = load_frag_a(QH, GDK, tm * 16, k0, lane);
        v16bf bb = load_frag_bt(ST, GDK, tn * 16, k0, lane);  // S[dk][dv] via S^T
        acc = wmma_bf16(a, bb, acc);
      }
#pragma unroll
      for (int s0 = 0; s0 < GC; s0 += 32) {
        v16bf a  = load_frag_a(Pm, GC, tm * 16, s0, lane);
        v16bf bb = load_frag_bt(VT, GC, tn * 16, s0, lane);
        acc = wmma_bf16(a, bb, acc);
      }
      const int trow0 = tm * 16 + ((lane >> 4) << 3);
      const int dv = tn * 16 + (lane & 15);
#pragma unroll
      for (int r = 0; r < 8; ++r) {
        const size_t m = (size_t)(b * GT + ck * GC + trow0 + r);
        obuf[m * GD + h * 256 + dv] = acc[r];
      }
    }
    __syncthreads();

    // ---- phase 4: S = S .* exp(Gend[dk]) + kc^T @ v, 256 tiles / 8 waves
    for (int i = 0; i < 32; ++i) {
      const int tt = wave * 32 + i;
      const int tmk = tt >> 4, tnv = tt & 15;
      v8f acc = {};
#pragma unroll
      for (int s0 = 0; s0 < GC; s0 += 32) {
        v16bf a  = load_frag_a(KCT, GC, tmk * 16, s0, lane);
        v16bf bb = load_frag_bt(VT, GC, tnv * 16, s0, lane);
        acc = wmma_bf16(a, bb, acc);
      }
      const int dk0 = tmk * 16 + ((lane >> 4) << 3);
      const int dv  = tnv * 16 + (lane & 15);
#pragma unroll
      for (int r = 0; r < 8; ++r) {
        const int dk = dk0 + r;
        const size_t idx = (size_t)dv * GDK + dk;
        ST[idx] = (__bf16)((float)ST[idx] * __expf(GE[dk]) + acc[r]);
      }
    }
    __syncthreads();
  }
}

// ---------------------------------------------------------------------------
// Gated RMSNorm: attn = rmsnorm(o) * rms_w * swish(g), bf16 out for Wo GEMM.
// ---------------------------------------------------------------------------
__global__ __launch_bounds__(256)
void gated_rmsnorm_kernel(const float* __restrict__ obuf,
                          const float* __restrict__ qkvg,
                          const float* __restrict__ rms_w,
                          __bf16* __restrict__ attn_bf) {
  __shared__ float red[256];
  const size_t row = blockIdx.x;           // row = m*H + h
  const size_t m = row >> 2;
  const int h = (int)(row & 3);
  const int d = threadIdx.x;
  const float ov = obuf[row * 256 + d];
  const float ms = block_reduce_sum256(ov * ov, red) * (1.0f / 256.0f);
  const float inv = rsqrtf(ms + 1e-5f);
  const float g = qkvg[m * 4096 + 3072 + h * 256 + d];
  const float sw = g / (1.0f + __expf(-g));          // g * sigmoid(g)
  attn_bf[m * GD + h * 256 + d] = (__bf16)(ov * inv * rms_w[d] * sw);
}

// ---------------------------------------------------------------------------
extern "C" void kernel_launch(void* const* d_in, const int* in_sizes, int n_in,
                              void* d_out, int out_size, void* d_ws, size_t ws_size,
                              hipStream_t stream) {
  const float* x     = (const float*)d_in[0];
  const float* ln1_w = (const float*)d_in[1];
  const float* ln1_b = (const float*)d_in[2];
  const float* Wq    = (const float*)d_in[3];
  const float* Wk    = (const float*)d_in[4];
  const float* Wv    = (const float*)d_in[5];
  const float* Wg    = (const float*)d_in[6];
  const float* Wgk1  = (const float*)d_in[7];
  const float* Wgk2  = (const float*)d_in[8];
  const float* bgk2  = (const float*)d_in[9];
  const float* rms_w = (const float*)d_in[10];
  const float* Wo    = (const float*)d_in[11];
  const float* ln2_w = (const float*)d_in[12];
  const float* ln2_b = (const float*)d_in[13];
  const float* W1    = (const float*)d_in[14];
  const float* b1    = (const float*)d_in[15];
  const float* W2    = (const float*)d_in[16];
  const float* b2    = (const float*)d_in[17];
  float* out = (float*)d_out;
  (void)in_sizes; (void)n_in; (void)out_size; (void)ws_size;

  char* ws = (char*)d_ws;
  auto alloc = [&](size_t bytes) -> char* {
    char* p = ws;
    ws += (bytes + 255) & ~(size_t)255;
    return p;
  };
  __bf16* xn_bf   = (__bf16*)alloc((size_t)GM * GD * 2);
  __bf16* wt_qkvg = (__bf16*)alloc((size_t)4096 * GD * 2);   // [4096][1024]
  __bf16* wt_o    = (__bf16*)alloc((size_t)GD * GD * 2);
  __bf16* wt_1    = (__bf16*)alloc((size_t)3072 * GD * 2);   // [3072][1024]
  __bf16* wt_2    = (__bf16*)alloc((size_t)GD * 3072 * 2);   // [1024][3072]
  float*  qkvg    = (float*)alloc((size_t)GM * 4096 * 4);
  float*  gkbuf   = (float*)alloc((size_t)GM * GD * 4);
  float*  obuf    = (float*)alloc((size_t)GM * GD * 4);
  __bf16* attn_bf = (__bf16*)alloc((size_t)GM * GD * 2);
  float*  x1      = (float*)alloc((size_t)GM * GD * 4);
  __bf16* xn2_bf  = (__bf16*)alloc((size_t)GM * GD * 2);
  __bf16* h_bf    = (__bf16*)alloc((size_t)GM * 3072 * 2);

  // 1. LN1 -> bf16
  layernorm_bf16_kernel<<<GM, 256, 0, stream>>>(x, ln1_w, ln1_b, xn_bf);

  // 2. Weight transpose+cast (q,k,v,g fused into one [4096][1024] buffer)
  const int tgrid_DD = (GD * GD + 255) / 256;
  transpose_cast_kernel<<<tgrid_DD, 256, 0, stream>>>(Wq, wt_qkvg + (size_t)0 * GD * GD, GD, GD);
  transpose_cast_kernel<<<tgrid_DD, 256, 0, stream>>>(Wk, wt_qkvg + (size_t)1 * GD * GD, GD, GD);
  transpose_cast_kernel<<<tgrid_DD, 256, 0, stream>>>(Wv, wt_qkvg + (size_t)2 * GD * GD, GD, GD);
  transpose_cast_kernel<<<tgrid_DD, 256, 0, stream>>>(Wg, wt_qkvg + (size_t)3 * GD * GD, GD, GD);
  transpose_cast_kernel<<<tgrid_DD, 256, 0, stream>>>(Wo, wt_o, GD, GD);
  transpose_cast_kernel<<<(GD * 3072 + 255) / 256, 256, 0, stream>>>(W1, wt_1, GD, 3072);
  transpose_cast_kernel<<<(3072 * GD + 255) / 256, 256, 0, stream>>>(W2, wt_2, 3072, GD);

  // 3. Low-rank forget gate
  gate_lowrank_kernel<<<GM, 256, 0, stream>>>(xn_bf, Wgk1, Wgk2, bgk2, gkbuf);

  // 4. Fused QKVG projection: [8192,4096] = xn_bf @ WT^T (TDM-staged weights)
  gemm_bf16_kernel<0, false, false, true, false>
      <<<dim3(GM / 128, 4096 / 64), 256, 0, stream>>>(
          xn_bf, wt_qkvg, nullptr, nullptr, qkvg, nullptr, 4096, GD);

  // 5. Chunkwise GLA scan (one WG per (b,h); ~265KB dynamic LDS on CDNA5)
  const size_t scan_lds = (size_t)(GDV * GDK + 3 * GC * GDK + GDV * GC + GC * GC) * 2
                        + GDK * 4;
  gla_scan_kernel<<<GB * GH, 256, scan_lds, stream>>>(qkvg, gkbuf, obuf);

  // 6. Gated RMSNorm -> bf16
  gated_rmsnorm_kernel<<<GM * GH, 256, 0, stream>>>(obuf, qkvg, rms_w, attn_bf);

  // 7. Output projection + residual: x1 = x + attn @ Wo
  gemm_bf16_kernel<0, false, true, true, false>
      <<<dim3(GM / 128, GD / 64), 256, 0, stream>>>(
          attn_bf, wt_o, nullptr, x, x1, nullptr, GD, GD);

  // 8. LN2 -> bf16
  layernorm_bf16_kernel<<<GM, 256, 0, stream>>>(x1, ln2_w, ln2_b, xn2_bf);

  // 9. MLP up: h = leaky_relu(xn2 @ W1 + b1)^2, bf16 out only
  gemm_bf16_kernel<1, true, false, false, true>
      <<<dim3(GM / 128, 3072 / 64), 256, 0, stream>>>(
          xn2_bf, wt_1, b1, nullptr, nullptr, h_bf, 3072, GD);

  // 10. MLP down + residual: out = x1 + h @ W2 + b2
  gemm_bf16_kernel<0, true, true, true, false>
      <<<dim3(GM / 128, GD / 64), 256, 0, stream>>>(
          h_bf, wt_2, b2, x1, out, nullptr, GD, 3072);
}
